// RADFE_28260884807864
// MI455X (gfx1250) — matrix operand order repaired and checked by reading, not compiled, not run on hardware
//
#include <hip/hip_runtime.h>

typedef __attribute__((ext_vector_type(16))) _Float16 v16h;
typedef __attribute__((ext_vector_type(8)))  float    v8f;

__device__ __forceinline__ float siluf(float v){ return v / (1.f + __expf(-v)); }
__device__ __forceinline__ float sigmf(float v){ return 1.f / (1.f + __expf(-v)); }
__device__ __forceinline__ float softplusf(float v){ return v > 20.f ? v : log1pf(__expf(v)); }

// CDNA5 async global->LDS copy (ASYNCcnt-tracked), ISA §15.18.3 op 98.
// Low 32 bits of a generic LDS pointer are the LDS byte address (ISA §10.2).
__device__ __forceinline__ void async_ld_b128(void* lds_ptr, const void* gptr) {
  unsigned l = (unsigned)(size_t)lds_ptr;
  unsigned long long g = (unsigned long long)(size_t)gptr;
  asm volatile("global_load_async_to_lds_b128 %0, %1, off"
               :: "v"(l), "v"(g) : "memory");
}
__device__ __forceinline__ void async_wait0() {
  asm volatile("s_wait_asynccnt 0" ::: "memory");
}

// ---------------------------------------------------------------------------
// Fused fast-mamba: LN(x over C) + lift + 32x mamba(dm=8,di=16,ds=16) + pool(L)
// grid = 512 blocks (one per (b,s)), block = 256 threads (8 lanes per channel).
// Lane j of channel c owns SSM rows i=2j,2j+1 (32 state floats in VGPRs).
// x tiles are staged into LDS with the CDNA5 async-DMA path.
// ---------------------------------------------------------------------------
__global__ __launch_bounds__(256) void k_fast(
    const float* __restrict__ x,
    const float* __restrict__ in_g, const float* __restrict__ in_b,
    const float* __restrict__ lift_w, const float* __restrict__ lift_b,
    const float* __restrict__ W_in, const float* __restrict__ conv_w,
    const float* __restrict__ conv_b, const float* __restrict__ W_xp,
    const float* __restrict__ W_dt, const float* __restrict__ b_dt,
    const float* __restrict__ A_log, const float* __restrict__ Dp,
    const float* __restrict__ W_out, const float* __restrict__ W_op,
    const float* __restrict__ b_op, float* __restrict__ chan)
{
  const int TL = 64;
  __shared__ __align__(16) float xt[64][32];
  __shared__ __align__(16) float ucs[32][16];
  __shared__ __align__(16) float zgs[32][16];
  __shared__ __align__(16) float sBm[32][16];
  __shared__ __align__(16) float sCm[32][16];
  __shared__ __align__(16) float sg [32][16];
  __shared__ __align__(16) float sWc[32][8][16];
  __shared__ float sbop[32][8];

  const int tid  = threadIdx.x;
  const int lane = tid & 31;
  const int wav  = tid >> 5;
  const int c    = tid >> 3;     // channel 0..31
  const int j    = tid & 7;      // sub-lane within channel
  const int bs   = blockIdx.x;   // b*256+s
  const int b    = bs >> 8;
  const int s    = bs & 255;

  // Wc = W_op @ W_out  (no nonlinearity between them in the reference)
  for (int idx = tid; idx < 32*8*16; idx += 256) {
    int cc = idx >> 7, m = (idx >> 4) & 7, i = idx & 15;
    float a = 0.f;
    #pragma unroll
    for (int mm = 0; mm < 8; ++mm)
      a += W_op[(cc*8+m)*8+mm] * W_out[(cc*8+mm)*16+i];
    sWc[cc][m][i] = a;
  }
  sbop[c][j] = b_op[c*8+j];

  // combined lift+W_in:  xz_o = xc * Axz + Bxz
  float Axz[4], Bxz[4];
  #pragma unroll
  for (int r = 0; r < 4; ++r) {
    int o = 4*j + r; float a = 0.f, bb = 0.f;
    #pragma unroll
    for (int d = 0; d < 8; ++d) {
      float wv = W_in[(c*32+o)*8+d];
      a += lift_w[c*8+d]*wv; bb += lift_b[c*8+d]*wv;
    }
    Axz[r] = a; Bxz[r] = bb;
  }
  float cw[4][4], cb[4];
  #pragma unroll
  for (int r = 0; r < 4; ++r) {
    int i = 4*(j&3)+r;
    cb[r] = conv_b[c*16+i];
    #pragma unroll
    for (int k = 0; k < 4; ++k) cw[r][k] = conv_w[(c*16+i)*4+k];
  }
  // W_xp row 0 (dt) + rows 1+4j..4+4j (Bm rows for j<4, Cm rows for j>=4)
  float wxp0[16], wxp[4][16];
  #pragma unroll
  for (int k = 0; k < 16; ++k) wxp0[k] = W_xp[(c*33)*16+k];
  #pragma unroll
  for (int r = 0; r < 4; ++r)
    #pragma unroll
    for (int k = 0; k < 16; ++k)
      wxp[r][k] = W_xp[(c*33 + 1 + 4*j + r)*16 + k];

  const int i0 = 2*j, i1 = 2*j+1;
  float A0[16], A1[16];
  #pragma unroll
  for (int ss = 0; ss < 16; ++ss) {
    A0[ss] = -__expf(A_log[(c*16+i0)*16+ss]);
    A1[ss] = -__expf(A_log[(c*16+i1)*16+ss]);
  }
  const float wdt0 = W_dt[c*16+i0], wdt1 = W_dt[c*16+i1];
  const float bdt0 = b_dt[c*16+i0], bdt1 = b_dt[c*16+i1];
  const float Dd0  = Dp[c*16+i0],  Dd1  = Dp[c*16+i1];
  const float gl = in_g[lane], blv = in_b[lane];

  float h0[16], h1[16];
  #pragma unroll
  for (int ss = 0; ss < 16; ++ss) { h0[ss] = 0.f; h1[ss] = 0.f; }
  float uh[4][3];
  #pragma unroll
  for (int r = 0; r < 4; ++r) { uh[r][0]=uh[r][1]=uh[r][2]=0.f; }
  float acc = 0.f;

  for (int l0 = 0; l0 < 512; l0 += TL) {
    __syncthreads();
    // ---- async-DMA the raw 64x32 tile into LDS (2 B128s per thread) ----
    {
      int q0 = tid, q1 = tid + 256;             // 512 x b128 covers the tile
      int ll0 = q0 >> 3, c40 = (q0 & 7) * 4;
      int ll1 = q1 >> 3, c41 = (q1 & 7) * 4;
      const float* g0 = x + (((size_t)(b*512 + l0 + ll0))*256 + s)*32 + c40;
      const float* g1 = x + (((size_t)(b*512 + l0 + ll1))*256 + s)*32 + c41;
      async_ld_b128(&xt[ll0][c40], g0);
      async_ld_b128(&xt[ll1][c41], g1);
      async_wait0();
    }
    __syncthreads();
    // ---- LayerNorm each row in place (one wave per row) ----
    for (int ll = wav; ll < TL; ll += 8) {
      float v = xt[ll][lane];
      float sum = v, sq = v*v;
      #pragma unroll
      for (int off = 16; off > 0; off >>= 1) {
        sum += __shfl_xor(sum, off, 32);
        sq  += __shfl_xor(sq,  off, 32);
      }
      float mn  = sum * (1.f/32.f);
      float var = sq  * (1.f/32.f) - mn*mn;
      xt[ll][lane] = (v - mn) * rsqrtf(var + 1e-5f) * gl + blv;
    }
    __syncthreads();

    for (int ll = 0; ll < TL; ++ll) {
      float xc = xt[ll][c];
      if (j < 4) {                                   // u path: conv + silu
        #pragma unroll
        for (int r = 0; r < 4; ++r) {
          float uo = fmaf(xc, Axz[r], Bxz[r]);
          float u4 = cb[r] + cw[r][0]*uh[r][0] + cw[r][1]*uh[r][1]
                   + cw[r][2]*uh[r][2] + cw[r][3]*uo;
          uh[r][0]=uh[r][1]; uh[r][1]=uh[r][2]; uh[r][2]=uo;
          ucs[c][4*j+r] = siluf(u4);
        }
      } else {                                       // gate path
        #pragma unroll
        for (int r = 0; r < 4; ++r) {
          float zo = fmaf(xc, Axz[r], Bxz[r]);
          zgs[c][4*(j-4)+r] = siluf(zo);
        }
      }
      __syncthreads();
      float uv[16];
      #pragma unroll
      for (int k = 0; k < 16; ++k) uv[k] = ucs[c][k];
      float dt_in = 0.f;
      #pragma unroll
      for (int k = 0; k < 16; ++k) dt_in = fmaf(uv[k], wxp0[k], dt_in);
      #pragma unroll
      for (int r = 0; r < 4; ++r) {
        float o = 0.f;
        #pragma unroll
        for (int k = 0; k < 16; ++k) o = fmaf(uv[k], wxp[r][k], o);
        if (j < 4) sBm[c][4*j+r] = o; else sCm[c][4*(j-4)+r] = o;
      }
      __syncthreads();
      float Bm[16], Cm[16];
      #pragma unroll
      for (int k = 0; k < 16; ++k) { Bm[k] = sBm[c][k]; Cm[k] = sCm[c][k]; }
      float u0 = ucs[c][i0], u1 = ucs[c][i1];
      float dt0 = softplusf(fmaf(dt_in, wdt0, bdt0));
      float dt1 = softplusf(fmaf(dt_in, wdt1, bdt1));
      float du0 = dt0*u0, du1 = dt1*u1;
      float y0 = 0.f, y1 = 0.f;
      #pragma unroll
      for (int ss = 0; ss < 16; ++ss) {
        h0[ss] = fmaf(h0[ss], __expf(dt0*A0[ss]), du0*Bm[ss]);
        y0 = fmaf(h0[ss], Cm[ss], y0);
        h1[ss] = fmaf(h1[ss], __expf(dt1*A1[ss]), du1*Bm[ss]);
        y1 = fmaf(h1[ss], Cm[ss], y1);
      }
      y0 = fmaf(u0, Dd0, y0); y1 = fmaf(u1, Dd1, y1);
      sg[c][i0] = y0 * zgs[c][i0];
      sg[c][i1] = y1 * zgs[c][i1];
      __syncthreads();
      float rs = sbop[c][j];
      #pragma unroll
      for (int k = 0; k < 16; ++k) rs = fmaf(sWc[c][j][k], sg[c][k], rs);
      acc += siluf(rs);
    }
  }
  chan[(size_t)bs*256 + c*8 + j] = acc * (1.f/512.f);
}

// ---------------------------------------------------------------------------
// WMMA f16 GEMM: C[M,N] = act(A[M,K] @ W[N,K]^T + bias). One wave per 32x64
// tile: 2 A-fragments x 4 B-fragments -> 8 WMMAs per 32-K step (each B frag
// reused 2x, each A frag 4x). Requires K%32==0, M%32==0, N%16==0.
// ---------------------------------------------------------------------------
__global__ __launch_bounds__(32) void k_gemm(
    const float* __restrict__ A, const float* __restrict__ W,
    const float* __restrict__ bias, float* __restrict__ C,
    int M, int N, int K, int act)
{
  const int lane = threadIdx.x;
  const int m0 = blockIdx.x * 32;
  const int nbase = blockIdx.y * 64;
  const int half = lane >> 4;
  const int l16 = lane & 15;
  int ntiles = (N - nbase) >> 4; if (ntiles > 4) ntiles = 4;
  v8f zero = {};
  v8f acc[2][4];
  #pragma unroll
  for (int a1 = 0; a1 < 2; ++a1)
    #pragma unroll
    for (int a2 = 0; a2 < 4; ++a2) acc[a1][a2] = zero;
  const float* arow0 = A + (size_t)(m0 + l16) * K;
  const float* arow1 = arow0 + (size_t)16 * K;
  for (int k0 = 0; k0 < K; k0 += 32) {
    v16h af0, af1;                                      // A-layout: k = e (+16 hi half)
    #pragma unroll
    for (int e = 0; e < 8; ++e) {
      af0[e] = (_Float16)arow0[k0 + e + (half ? 16 : 0)];
      af1[e] = (_Float16)arow1[k0 + e + (half ? 16 : 0)];
    }
    #pragma unroll
    for (int e = 8; e < 16; ++e) {
      af0[e] = (_Float16)arow0[k0 + e + (half ? 16 : 0)];
      af1[e] = (_Float16)arow1[k0 + e + (half ? 16 : 0)];
    }
    #pragma unroll
    for (int nt = 0; nt < 4; ++nt) {
      if (nt >= ntiles) break;
      const float* wrow = W + (size_t)(nbase + nt*16 + l16) * K + k0 + (half ? 16 : 0);
      v16h bf;                                          // B-layout: k = e (+16 hi half)
      #pragma unroll
      for (int e = 0; e < 16; ++e) bf[e] = (_Float16)wrow[e];
      acc[0][nt] = __builtin_amdgcn_wmma_f32_16x16x32_f16(false, af0, false, bf,
                                                          (short)0, acc[0][nt], false, false);
      acc[1][nt] = __builtin_amdgcn_wmma_f32_16x16x32_f16(false, af1, false, bf,
                                                          (short)0, acc[1][nt], false, false);
    }
  }
  #pragma unroll
  for (int mt = 0; mt < 2; ++mt)
    #pragma unroll
    for (int nt = 0; nt < 4; ++nt) {
      if (nt >= ntiles) continue;
      int n = nbase + nt*16 + l16;
      float bv = bias ? bias[n] : 0.f;
      v8f a = acc[mt][nt];
      #pragma unroll
      for (int r = 0; r < 8; ++r) {
        int m = m0 + mt*16 + r + half*8;                // C/D layout
        float v = a[r] + bv;
        if (act == 1) v = siluf(v);
        else if (act == 2) v = sigmf(v);
        C[(size_t)m*N + n] = v;
      }
    }
}

// LayerNorm over last dim 256 (one row per block)
__global__ __launch_bounds__(256) void k_ln256(
    const float* __restrict__ in, const float* __restrict__ g,
    const float* __restrict__ b, float* __restrict__ out)
{
  __shared__ float rs[8], rq[8];
  int r = blockIdx.x, t = threadIdx.x;
  float v = in[(size_t)r*256 + t];
  float sum = v, sq = v*v;
  #pragma unroll
  for (int off = 16; off > 0; off >>= 1) {
    sum += __shfl_xor(sum, off, 32); sq += __shfl_xor(sq, off, 32);
  }
  if ((t & 31) == 0) { rs[t>>5] = sum; rq[t>>5] = sq; }
  __syncthreads();
  float ts = 0.f, tq = 0.f;
  #pragma unroll
  for (int wv = 0; wv < 8; ++wv) { ts += rs[wv]; tq += rq[wv]; }
  float mn  = ts * (1.f/256.f);
  float var = tq * (1.f/256.f) - mn*mn;
  out[(size_t)r*256 + t] = (v - mn) * rsqrtf(var + 1e-5f) * g[t] + b[t];
}

// slow mamba: depthwise causal conv K=4 over L + silu. rows=(b*256+l), 512 ch
__global__ void k_slow_conv(const float* __restrict__ xz, const float* __restrict__ cwv,
                            const float* __restrict__ cbv, float* __restrict__ uc)
{
  int idx = blockIdx.x*256 + threadIdx.x;      // 512*512
  int i = idx & 511; int r = idx >> 9; int l = r & 255; int bb = r >> 8;
  float a = cbv[i];
  #pragma unroll
  for (int k = 0; k < 4; ++k) {
    int lk = l - 3 + k;
    if (lk >= 0) a = fmaf(xz[(size_t)(bb*256+lk)*1024 + i], cwv[i*4+k], a);
  }
  uc[(size_t)r*512 + i] = siluf(a);
}

// dt = softplus(x_dbl[:, :16] @ W_dt^T + b_dt)
__global__ void k_slow_dt(const float* __restrict__ xdbl, const float* __restrict__ Wdt,
                          const float* __restrict__ bdt, float* __restrict__ dt)
{
  int idx = blockIdx.x*256 + threadIdx.x;      // 512*512
  int i = idx & 511, r = idx >> 9;
  float a = bdt[i];
  #pragma unroll
  for (int k = 0; k < 16; ++k) a = fmaf(xdbl[(size_t)r*80 + k], Wdt[i*16+k], a);
  dt[(size_t)r*512 + i] = softplusf(a);
}

// slow SSM scan: one thread per (b,i), 32-wide state in VGPRs, B/C via LDS
__global__ __launch_bounds__(256) void k_slow_scan(
    const float* __restrict__ xdbl, const float* __restrict__ dt,
    const float* __restrict__ uc, const float* __restrict__ xz,
    const float* __restrict__ A_log, const float* __restrict__ Dp,
    float* __restrict__ g)
{
  __shared__ float sB[32], sC[32];
  int bb = blockIdx.x >> 1;
  int i  = ((blockIdx.x & 1) << 8) + threadIdx.x;
  float Av[32], h[32];
  #pragma unroll
  for (int s = 0; s < 32; ++s) { Av[s] = -__expf(A_log[i*32+s]); h[s] = 0.f; }
  float Di = Dp[i];
  for (int l = 0; l < 256; ++l) {
    int r = bb*256 + l;
    __syncthreads();
    if (threadIdx.x < 32)      sB[threadIdx.x]    = xdbl[(size_t)r*80 + 16 + threadIdx.x];
    else if (threadIdx.x < 64) sC[threadIdx.x-32] = xdbl[(size_t)r*80 + 48 + (threadIdx.x-32)];
    __syncthreads();
    float dtv = dt[(size_t)r*512 + i];
    float uv  = uc[(size_t)r*512 + i];
    float du = dtv*uv, y = 0.f;
    #pragma unroll
    for (int s = 0; s < 32; ++s) {
      h[s] = fmaf(h[s], __expf(dtv*Av[s]), du*sB[s]);
      y = fmaf(h[s], sC[s], y);
    }
    y = fmaf(uv, Di, y);
    float zg = xz[(size_t)r*1024 + 512 + i];
    g[(size_t)r*512 + i] = y * siluf(zg);
  }
}

__global__ void k_cat(const float* __restrict__ hh, const float* __restrict__ yb,
                      float* __restrict__ cat)
{
  int idx = blockIdx.x*256 + threadIdx.x;      // 512*512
  int cc = idx & 511, r = idx >> 9;
  float v = (cc < 256) ? hh[(size_t)r*256 + cc] + yb[(size_t)r*256 + cc]
                       : hh[(size_t)r*256 + (cc-256)];
  cat[(size_t)r*512 + cc] = v;
}

__global__ void k_im2col(const float* __restrict__ proj, float* __restrict__ col)
{
  int idx = blockIdx.x*256 + threadIdx.x;
  if (idx >= 2048*2304) return;
  int k = idx % 2304; int m = idx / 2304;
  int kw = k % 3, kh = (k/3) % 3, ci = k/9;
  int xx = m & 31, yy = (m >> 5) & 31, bb = m >> 10;
  int sy = yy + kh - 1, sx = xx + kw - 1;
  float v = 0.f;
  if (sy >= 0 && sy < 32 && sx >= 0 && sx < 32)
    v = proj[(size_t)(bb*256 + ci)*1024 + sy*32 + sx];
  col[(size_t)m*2304 + k] = v;
}

__global__ void k_bnsilu_t1(const float* __restrict__ c1out,
    const float* __restrict__ bg, const float* __restrict__ bb_,
    const float* __restrict__ bm, const float* __restrict__ bv,
    float* __restrict__ t1)
{
  int idx = blockIdx.x*256 + threadIdx.x;      // 2*32*32*32
  int xx = idx & 31, yy = (idx>>5)&31, co = (idx>>10)&31, b2 = idx>>15;
  float v = siluf(c1out[(size_t)(b2*1024 + yy*32 + xx)*32 + co]);
  v = (v - bm[co]) * rsqrtf(bv[co] + 1e-5f) * bg[co] + bb_[co];
  t1[idx] = v;
}

// generic 3x3 SAME conv; mode 0:+bias, 1:silu(+bias), 2:sigmoid(+bias),
// 3:bn(silu(conv)), 4:bn(conv). out_cap/co_off let us write into d_out slices.
__global__ void k_conv3x3(const float* __restrict__ in, const float* __restrict__ w,
    const float* __restrict__ bias,
    const float* __restrict__ bg, const float* __restrict__ bb_,
    const float* __restrict__ bm, const float* __restrict__ bv,
    float* __restrict__ out,
    int B2, int Cin, int Cout, int H, int W2, int mode, int out_cap, int co_off)
{
  int idx = blockIdx.x*256 + threadIdx.x;
  if (idx >= B2*Cout*H*W2) return;
  int xx = idx % W2; int t = idx / W2;
  int yy = t % H; t /= H;
  int co = t % Cout; int b2 = t / Cout;
  float a = bias ? bias[co] : 0.f;
  for (int ci = 0; ci < Cin; ++ci) {
    const float* ip = in + ((size_t)(b2*Cin + ci)*H)*W2;
    const float* wp = w + ((size_t)(co*Cin + ci))*9;
    #pragma unroll
    for (int kh = 0; kh < 3; ++kh) {
      int sy = yy + kh - 1;
      if (sy < 0 || sy >= H) continue;
      #pragma unroll
      for (int kw = 0; kw < 3; ++kw) {
        int sx = xx + kw - 1;
        if (sx < 0 || sx >= W2) continue;
        a = fmaf(ip[(size_t)sy*W2 + sx], wp[kh*3 + kw], a);
      }
    }
  }
  if (mode == 1) a = siluf(a);
  else if (mode == 2) a = sigmf(a);
  else if (mode == 3 || mode == 4) {
    if (mode == 3) a = siluf(a);
    a = (a - bm[co]) * rsqrtf(bv[co] + 1e-5f) * bg[co] + bb_[co];
  }
  out[(((size_t)b2*out_cap + co_off + co)*H + yy)*W2 + xx] = a;
}

__global__ void k_resize(const float* __restrict__ in, float* __restrict__ out,
                         int B2, int C2, int Hi, int Wi, int Ho, int Wo)
{
  int idx = blockIdx.x*256 + threadIdx.x;
  if (idx >= B2*C2*Ho*Wo) return;
  int xx = idx % Wo; int t = idx / Wo;
  int yy = t % Ho; t /= Ho;
  int cc = t % C2; int b2 = t / C2;
  float fy = (yy + 0.5f) * ((float)Hi / Ho) - 0.5f;
  float fx = (xx + 0.5f) * ((float)Wi / Wo) - 0.5f;
  int y0 = (int)floorf(fy), x0 = (int)floorf(fx);
  float wy = fy - y0, wx = fx - x0;
  int y0c = min(max(y0,0), Hi-1), y1c = min(max(y0+1,0), Hi-1);
  int x0c = min(max(x0,0), Wi-1), x1c = min(max(x0+1,0), Wi-1);
  const float* ip = in + (size_t)(b2*C2 + cc)*Hi*Wi;
  float v = (1.f-wy)*((1.f-wx)*ip[y0c*Wi+x0c] + wx*ip[y0c*Wi+x1c])
          +       wy*((1.f-wx)*ip[y1c*Wi+x0c] + wx*ip[y1c*Wi+x1c]);
  out[idx] = v;
}

__global__ void k_smean(const float* __restrict__ proj, float* __restrict__ sm)
{
  int idx = blockIdx.x*256 + threadIdx.x;      // 2048
  if (idx >= 2048) return;
  int p = idx & 1023, b2 = idx >> 10;
  float a = 0.f;
  for (int s = 0; s < 256; ++s) a += proj[(size_t)(b2*256 + s)*1024 + p];
  sm[idx] = a * (1.f/256.f);
}

// ---------------------------------------------------------------------------
extern "C" void kernel_launch(void* const* d_in, const int* in_sizes, int n_in,
                              void* d_out, int out_size, void* d_ws, size_t ws_size,
                              hipStream_t stream)
{
  (void)in_sizes; (void)n_in; (void)out_size; (void)ws_size;
  #define IN(i) ((const float*)d_in[(i)])
  const float* X      = IN(0);
  const float* in_g   = IN(1);  const float* in_b   = IN(2);
  const float* lift_w = IN(3);  const float* lift_b = IN(4);
  const float* fWin = IN(5),  *fcw = IN(6),  *fcb = IN(7),  *fWxp = IN(8);
  const float* fWdt = IN(9),  *fbdt = IN(10), *fAlog = IN(11), *fDD = IN(12);
  const float* fWout = IN(13), *fWop = IN(14), *fbop = IN(15);
  const float* pl_g = IN(16), *pl_b = IN(17);
  const float* sl_w = IN(18), *sl_b = IN(19), *sn_g = IN(20), *sn_b = IN(21);
  const float* sWin = IN(22), *scw = IN(23), *scb = IN(24), *sWxp = IN(25);
  const float* sWdt = IN(26), *sbdt = IN(27), *sAlog = IN(28), *sDD = IN(29);
  const float* sWout = IN(30), *sWop = IN(31), *sbop2 = IN(32);
  const float* projw = IN(33), *projb = IN(34);
  const float* c1w = IN(35);
  const float *bn1g=IN(36), *bn1b=IN(37), *bn1m=IN(38), *bn1v=IN(39);
  const float* c2w = IN(40);
  const float *bn2g=IN(41), *bn2b=IN(42), *bn2m=IN(43), *bn2v=IN(44);
  const float* c3w = IN(45);
  const float *bn3g=IN(46), *bn3b=IN(47), *bn3m=IN(48), *bn3v=IN(49);
  const float* c4w = IN(50);
  const float *bn4g=IN(51), *bn4b=IN(52), *bn4m=IN(53), *bn4v=IN(54);
  const float *clsw=IN(55), *clsb=IN(56), *regw=IN(57), *regb=IN(58);
  const float *c0w=IN(59), *c0b=IN(60), *c11w=IN(61), *c11b=IN(62);
  const float *c12w=IN(63), *c12b=IN(64), *c21w=IN(65), *c21b=IN(66);
  const float *c22w=IN(67), *c22b=IN(68);

  float* Wsp = (float*)d_ws;
  size_t off = 0;
  auto alc = [&](size_t n){ float* p = Wsp + off; off += n; return p; };
  float* chan = alc(131072);   float* cl   = alc(131072);
  float* s1   = alc(131072);   float* hb   = alc(131072);
  float* xz   = alc(524288);   float* ucb  = alc(262144);
  float* xdbl = alc(40960);    float* dtb  = alc(262144);
  float* gb   = alc(262144);   float* o1   = alc(131072);
  float* yb   = alc(131072);   float* cat  = alc(262144);
  float* proj = alc(524288);   float* col  = alc(4718592ULL);
  float* c1o  = alc(65536);    float* t1   = alc(65536);
  float* t2   = alc(32768);    float* t2r  = alc(229376);
  float* t3   = alc(114688);   float* t3r  = alc(458752);
  float* t4   = alc(458752);   float* sm   = alc(2048);
  float* s0b  = alc(16384);    float* s0r  = alc(65536);
  float* sg1b = alc(65536);    float* sg2b = alc(32768);
  float* sg2r = alc(131072);   float* sg3b = alc(131072);
  float* outD = (float*)d_out;           // det: 172032 floats, seg follows

  // ---- fast mamba bank (fused LN+lift+scan+pool, async-DMA staging) ----
  k_fast<<<512, 256, 0, stream>>>(X, in_g, in_b, lift_w, lift_b,
      fWin, fcw, fcb, fWxp, fWdt, fbdt, fAlog, fDD, fWout, fWop, fbop, chan);

  // ---- slow path (WMMA GEMMs on 512 rows, 32x64 per wave) ----
  k_ln256<<<512, 256, 0, stream>>>(chan, pl_g, pl_b, cl);
  k_gemm<<<dim3(16, 4), 32, 0, stream>>>(cl, sl_w, sl_b, s1, 512, 256, 256, 1);
  k_ln256<<<512, 256, 0, stream>>>(s1, sn_g, sn_b, hb);
  k_gemm<<<dim3(16,16), 32, 0, stream>>>(hb, sWin, nullptr, xz, 512,1024, 256, 0);
  k_slow_conv<<<1024, 256, 0, stream>>>(xz, scw, scb, ucb);
  k_gemm<<<dim3(16, 2), 32, 0, stream>>>(ucb, sWxp, nullptr, xdbl, 512, 80, 512, 0);
  k_slow_dt<<<1024, 256, 0, stream>>>(xdbl, sWdt, sbdt, dtb);
  k_slow_scan<<<4, 256, 0, stream>>>(xdbl, dtb, ucb, xz, sAlog, sDD, gb);
  k_gemm<<<dim3(16, 4), 32, 0, stream>>>(gb, sWout, nullptr, o1, 512, 256, 512, 0);
  k_gemm<<<dim3(16, 4), 32, 0, stream>>>(o1, sWop, sbop2, yb, 512, 256, 256, 1);
  k_cat<<<1024, 256, 0, stream>>>(hb, yb, cat);
  k_gemm<<<dim3(16,16), 32, 0, stream>>>(cat, projw, projb, proj, 512,1024, 512, 0);

  // ---- detection head ----
  k_im2col<<<18432, 256, 0, stream>>>(proj, col);
  k_gemm<<<dim3(64, 1), 32, 0, stream>>>(col, c1w, nullptr, c1o, 2048, 32, 2304, 0);
  k_bnsilu_t1<<<256, 256, 0, stream>>>(c1o, bn1g, bn1b, bn1m, bn1v, t1);
  k_conv3x3<<<128, 256, 0, stream>>>(t1, c2w, nullptr, bn2g,bn2b,bn2m,bn2v, t2,
                                     2,32,16,32,32, 3, 16, 0);
  k_resize<<<896, 256, 0, stream>>>(t2, t2r, 2,16,32,32,64,112);
  k_conv3x3<<<448, 256, 0, stream>>>(t2r, c3w, nullptr, bn3g,bn3b,bn3m,bn3v, t3,
                                     2,16,8,64,112, 3, 8, 0);
  k_resize<<<1792, 256, 0, stream>>>(t3, t3r, 2,8,64,112,128,224);
  k_conv3x3<<<1792, 256, 0, stream>>>(t3r, c4w, nullptr, bn4g,bn4b,bn4m,bn4v, t4,
                                      2,8,8,128,224, 4, 8, 0);
  k_conv3x3<<<224, 256, 0, stream>>>(t4, clsw, clsb, nullptr,nullptr,nullptr,nullptr,
                                     outD, 2,8,1,128,224, 2, 3, 0);
  k_conv3x3<<<448, 256, 0, stream>>>(t4, regw, regb, nullptr,nullptr,nullptr,nullptr,
                                     outD, 2,8,2,128,224, 0, 3, 1);

  // ---- segmentation head ----
  k_smean<<<8, 256, 0, stream>>>(proj, sm);
  k_conv3x3<<<64, 256, 0, stream>>>(sm, c0w, c0b, nullptr,nullptr,nullptr,nullptr,
                                    s0b, 2,1,8,32,32, 1, 8, 0);
  k_resize<<<256, 256, 0, stream>>>(s0b, s0r, 2,8,32,32,64,64);
  k_conv3x3<<<256, 256, 0, stream>>>(s0r, c11w, c11b, nullptr,nullptr,nullptr,nullptr,
                                     sg1b, 2,8,8,64,64, 1, 8, 0);
  k_conv3x3<<<128, 256, 0, stream>>>(sg1b, c12w, c12b, nullptr,nullptr,nullptr,nullptr,
                                     sg2b, 2,8,4,64,64, 1, 4, 0);
  k_resize<<<512, 256, 0, stream>>>(sg2b, sg2r, 2,4,64,64,128,128);
  k_conv3x3<<<512, 256, 0, stream>>>(sg2r, c21w, c21b, nullptr,nullptr,nullptr,nullptr,
                                     sg3b, 2,4,4,128,128, 1, 4, 0);
  k_conv3x3<<<128, 256, 0, stream>>>(sg3b, c22w, c22b, nullptr,nullptr,nullptr,nullptr,
                                     outD + 172032, 2,4,1,128,128, 0, 1, 0);
  #undef IN
}
